// MyGNN2_49555332661653
// MI455X (gfx1250) — compile-verified, matrix operand order checked by reading, not run on hardware
//
#include <hip/hip_runtime.h>

// ---------------------------------------------------------------------------
// CDNA5 (gfx1250) wave32 WMMA GNN pipeline.
// Heavy GEMMs -> v_wmma_f32_16x16x32_bf16 (NT layout, K-contiguous operands).
// Dominant GedMatrix kernel stages A-tiles through LDS with
// GLOBAL_LOAD_ASYNC_TO_LDS_B128 (double buffered, s_wait_asynccnt-gated).
// ---------------------------------------------------------------------------

typedef __attribute__((ext_vector_type(16))) __bf16 v16bf;
typedef __attribute__((ext_vector_type(8)))  float  v8f;

#define DEVINL static __device__ __forceinline__

DEVINL __bf16 f2bf(float f) {
  union { float f; unsigned u; } a; a.f = f;
  unsigned r = a.u + 0x7FFFu + ((a.u >> 16) & 1u);   // round-to-nearest-even
  unsigned short h = (unsigned short)(r >> 16);
  __bf16 b; __builtin_memcpy(&b, &h, 2); return b;
}

union FragU { v16bf f; uint4 u[2]; };

// NT fragment gather (global): lane l%16 -> row in tile, l/16 -> K-half.
DEVINL v16bf load_frag_nt(const __bf16* mat, int row0, int k0, int ld, int lane) {
  int lm = lane & 15, lh = lane >> 4;
  const __bf16* p = mat + (size_t)(row0 + lm) * (size_t)ld + (size_t)(k0 + lh * 8);
  FragU r;
  r.u[0] = *(const uint4*)(p);        // K = base .. base+7
  r.u[1] = *(const uint4*)(p + 16);   // K = base+16 .. base+23
  return r.f;
}

// Same gather from an LDS tile with padded row stride (136 bf16 = 68 dwords;
// 68 mod 64 = 4 -> the 16 lanes of a ds_load_b128 gather are bank-conflict-free).
#define LDS_PITCH 136
DEVINL v16bf load_frag_lds(const __bf16* tile, int row0, int k0, int lane) {
  int lm = lane & 15, lh = lane >> 4;
  const __bf16* p = tile + (row0 + lm) * LDS_PITCH + k0 + lh * 8;
  FragU r;
  r.u[0] = *(const uint4*)(p);
  r.u[1] = *(const uint4*)(p + 16);
  return r.f;
}

DEVINL v8f wmma_bf16(v16bf a, v16bf b, v8f c) {
  return __builtin_amdgcn_wmma_f32_16x16x32_bf16(false, a, false, b, (short)0, c,
                                                 false, false);
}

DEVINL v8f vzero8() { v8f z; for (int i = 0; i < 8; ++i) z[i] = 0.f; return z; }

// Async global -> LDS copy of 16 bytes per lane (ASYNCcnt-tracked).
DEVINL void async_ld_b128(unsigned lds_addr, const void* gaddr) {
  asm volatile("global_load_async_to_lds_b128 %0, %1, off"
               :: "v"(lds_addr), "v"(gaddr) : "memory");
}

// ---------------------------------------------------------------------------
// Generic NT GEMM:  C[M,N] = act(A[M,K] * B[N,K]^T + bias[N])
// block = 256 threads = 8 waves (2x4), block tile 64x128, wave tile 32x32.
// ---------------------------------------------------------------------------
__global__ void k_gemm_nt(const __bf16* __restrict__ A, const __bf16* __restrict__ B,
                          const float* __restrict__ bias,
                          float* __restrict__ C, __bf16* __restrict__ Cbf,
                          int M, int N, int K, int lda, int ldb, int ldc, int act)
{
  int lane = threadIdx.x & 31;
  int wave = threadIdx.x >> 5;
  int m0 = blockIdx.y * 64 + (wave >> 2) * 32;
  int n0 = blockIdx.x * 128 + (wave & 3) * 32;
  if (m0 >= M || n0 >= N) return;

  v8f acc[2][2];
  acc[0][0] = vzero8(); acc[0][1] = vzero8();
  acc[1][0] = vzero8(); acc[1][1] = vzero8();

  for (int kb = 0; kb < K; kb += 32) {
    v16bf a0 = load_frag_nt(A, m0,      kb, lda, lane);
    v16bf a1 = load_frag_nt(A, m0 + 16, kb, lda, lane);
    v16bf b0 = load_frag_nt(B, n0,      kb, ldb, lane);
    v16bf b1 = load_frag_nt(B, n0 + 16, kb, ldb, lane);
    acc[0][0] = wmma_bf16(a0, b0, acc[0][0]);
    acc[0][1] = wmma_bf16(a0, b1, acc[0][1]);
    acc[1][0] = wmma_bf16(a1, b0, acc[1][0]);
    acc[1][1] = wmma_bf16(a1, b1, acc[1][1]);
  }

  int lm = lane & 15, lh = lane >> 4;
  for (int i = 0; i < 2; ++i)
    for (int j = 0; j < 2; ++j) {
      int col = n0 + j * 16 + lm;
      float bv = bias ? bias[col] : 0.f;
      for (int r = 0; r < 8; ++r) {
        int row = m0 + i * 16 + r + lh * 8;
        float v = acc[i][j][r] + bv;
        if (act == 1 && v < 0.f) v = 0.f;
        size_t o = (size_t)row * (size_t)ldc + (size_t)col;
        if (C)   C[o]   = v;
        if (Cbf) Cbf[o] = f2bf(v);
      }
    }
}

// ---------------------------------------------------------------------------
// GedMatrix:
//   out[m,n] = postScale * sigmoid( sum_k v[k] * relu( Tk[m, k*128:..] . E2[n,:] ) )
// TK: n1 x (16*128) bf16, E2: n2 x 128 bf16.
// Block 64(m) x 128(n); TK 64x128 tiles double-buffered through LDS via
// async global->LDS loads (4x VMEM traffic reduction: tile shared by 4 waves);
// E2 (B) fragments hoisted in registers across the whole k loop.
// Grids are exact: n1 % 64 == 0, n2 % 128 == 0 (no guards; kernel has barriers).
// ---------------------------------------------------------------------------
__global__ void k_ged(const __bf16* __restrict__ TK, const __bf16* __restrict__ E2,
                      const float* __restrict__ v, float* __restrict__ out,
                      int n1, int n2, float postScale)
{
  __shared__ __bf16 smA[2][64 * LDS_PITCH];
  const int t = threadIdx.x;
  const int lane = t & 31;
  const int wave = t >> 5;
  const int row0 = blockIdx.y * 64;                 // block's M origin
  const int mw   = (wave >> 2) * 32;                // wave's M offset in tile
  const int n0   = blockIdx.x * 128 + (wave & 3) * 32;

  // stage a 64x128 bf16 TK tile (for matrix k) into LDS buffer `buf`
  auto stage = [&](int k, int buf) {
    unsigned base = (unsigned)(size_t)(&smA[buf][0]);
    const int kcol = k * 128;
#pragma unroll
    for (int j = 0; j < 4; ++j) {
      int q   = t + j * 256;          // 0..1023 : 64 rows x 16 segments
      int r   = q >> 4;
      int seg = q & 15;
      const __bf16* g = TK + (size_t)(row0 + r) * 2048 + (size_t)(kcol + seg * 8);
      unsigned l = base + (unsigned)(r * LDS_PITCH + seg * 8) * 2u;
      async_ld_b128(l, (const void*)g);
    }
  };

  // hoist all B fragments (reused by all 16 k-matrices)
  v16bf bB[4][2];
  for (int kb = 0; kb < 4; ++kb)
    for (int j = 0; j < 2; ++j)
      bB[kb][j] = load_frag_nt(E2, n0 + j * 16, kb * 32, 128, lane);

  float accO[2][2][8];
  for (int i = 0; i < 2; ++i)
    for (int j = 0; j < 2; ++j)
      for (int r = 0; r < 8; ++r) accO[i][j][r] = 0.f;

  stage(0, 0);
  for (int k = 0; k < 16; ++k) {
    if (k < 15) {
      stage(k + 1, (k + 1) & 1);
      asm volatile("s_wait_asynccnt 0x4" ::: "memory");   // tile k landed
    } else {
      asm volatile("s_wait_asynccnt 0x0" ::: "memory");
    }
    __syncthreads();                                      // tile visible to WG

    const __bf16* tile = smA[k & 1];
    v8f acc[2][2];
    acc[0][0] = vzero8(); acc[0][1] = vzero8();
    acc[1][0] = vzero8(); acc[1][1] = vzero8();
#pragma unroll
    for (int kb = 0; kb < 4; ++kb) {
      v16bf a0 = load_frag_lds(tile, mw,      kb * 32, lane);
      v16bf a1 = load_frag_lds(tile, mw + 16, kb * 32, lane);
      acc[0][0] = wmma_bf16(a0, bB[kb][0], acc[0][0]);
      acc[0][1] = wmma_bf16(a0, bB[kb][1], acc[0][1]);
      acc[1][0] = wmma_bf16(a1, bB[kb][0], acc[1][0]);
      acc[1][1] = wmma_bf16(a1, bB[kb][1], acc[1][1]);
    }
    float vk = v[k];
    for (int i = 0; i < 2; ++i)
      for (int j = 0; j < 2; ++j)
        for (int r = 0; r < 8; ++r) {
          float x = acc[i][j][r];
          accO[i][j][r] += vk * (x > 0.f ? x : 0.f);
        }
    __syncthreads();    // all waves done reading before buffer is overwritten
  }

  int lm = lane & 15, lh = lane >> 4;
  for (int i = 0; i < 2; ++i)
    for (int j = 0; j < 2; ++j)
      for (int r = 0; r < 8; ++r) {
        int row = row0 + mw + i * 16 + r + lh * 8;
        int col = n0 + j * 16 + lm;
        float s = 1.f / (1.f + expf(-accO[i][j][r]));
        out[(size_t)row * (size_t)n2 + (size_t)col] = s * postScale;
      }
}

// ----------------------------- small kernels --------------------------------

__global__ void k_zero(float* p, long n) {
  long i = (long)blockIdx.x * blockDim.x + threadIdx.x;
  if (i < n) p[i] = 0.f;
}

__global__ void k_f32_to_bf16(const float* s, __bf16* d, long n) {
  long i = (long)blockIdx.x * blockDim.x + threadIdx.x;
  if (i < n) d[i] = f2bf(s[i]);
}

// WT[b][c][r] = W[b][r][c]  (weights -> NT operand, bf16)
__global__ void k_transpose_bf16(const float* W, __bf16* WT, int rows, int cols, int batch) {
  long total = (long)batch * rows * cols;
  long i = (long)blockIdx.x * blockDim.x + threadIdx.x;
  if (i >= total) return;
  int b = (int)(i / ((long)rows * cols));
  long rem = i % ((long)rows * cols);
  int r = (int)(rem / cols), c = (int)(rem % cols);
  WT[(size_t)b * rows * cols + (size_t)c * rows + r] = f2bf(W[i]);
}

__global__ void k_scatter(const float* __restrict__ X, const int* __restrict__ ei,
                          float* __restrict__ AGG, int E, int F) {
  long i = (long)blockIdx.x * blockDim.x + threadIdx.x;
  if (i >= (long)E * F) return;
  int e = (int)(i / F), f = (int)(i % F);
  int s = ei[e], d = ei[E + e];
  atomicAdd(&AGG[(size_t)d * F + f], X[(size_t)s * F + f]);
}

__global__ void k_combine(const float* X, const float* AGG, const float* epsP,
                          __bf16* XBF, long n) {
  long i = (long)blockIdx.x * blockDim.x + threadIdx.x;
  if (i >= n) return;
  float e = 1.0f + epsP[0];
  XBF[i] = f2bf(e * X[i] + AGG[i]);
}

__global__ void k_bn_stats(const float* __restrict__ H, float* __restrict__ stats,
                           int N, int C) {
  int c = blockIdx.x;
  __shared__ float ss[256], sq[256];
  float s = 0.f, q = 0.f;
  for (int r = threadIdx.x; r < N; r += blockDim.x) {
    float x = H[(size_t)r * C + c]; s += x; q += x * x;
  }
  ss[threadIdx.x] = s; sq[threadIdx.x] = q; __syncthreads();
  for (int w = 128; w > 0; w >>= 1) {
    if ((int)threadIdx.x < w) { ss[threadIdx.x] += ss[threadIdx.x + w];
                                sq[threadIdx.x] += sq[threadIdx.x + w]; }
    __syncthreads();
  }
  if (threadIdx.x == 0) { stats[c] = ss[0]; stats[C + c] = sq[0]; }
}

__global__ void k_bn_apply(const float* H, const float* stats, const float* gamma,
                           const float* beta, float* Y, __bf16* Ybf,
                           long n, int C, float invN, int act) {
  long i = (long)blockIdx.x * blockDim.x + threadIdx.x;
  if (i >= n) return;
  int c = (int)(i % C);
  float mu  = stats[c] * invN;
  float var = stats[C + c] * invN - mu * mu;
  float x = (H[i] - mu) * rsqrtf(var + 1e-5f) * gamma[c] + beta[c];
  if (act && x < 0.f) x = 0.f;
  Y[i] = x;
  if (Ybf) Ybf[i] = f2bf(x);
}

__global__ void k_lse_row(float* LA, int R, int C) {
  int row = blockIdx.x;
  float* p = LA + (size_t)row * C;
  __shared__ float sh[256];
  float mx = -3.402823e38f;
  for (int c = threadIdx.x; c < C; c += blockDim.x) mx = fmaxf(mx, p[c]);
  sh[threadIdx.x] = mx; __syncthreads();
  for (int w = 128; w > 0; w >>= 1) {
    if ((int)threadIdx.x < w) sh[threadIdx.x] = fmaxf(sh[threadIdx.x], sh[threadIdx.x + w]);
    __syncthreads();
  }
  mx = sh[0]; __syncthreads();
  float s = 0.f;
  for (int c = threadIdx.x; c < C; c += blockDim.x) s += expf(p[c] - mx);
  sh[threadIdx.x] = s; __syncthreads();
  for (int w = 128; w > 0; w >>= 1) {
    if ((int)threadIdx.x < w) sh[threadIdx.x] += sh[threadIdx.x + w];
    __syncthreads();
  }
  float lse = mx + logf(sh[0]);
  for (int c = threadIdx.x; c < C; c += blockDim.x) p[c] -= lse;
}

__global__ void k_lse_col(float* LA, int R, int C) {
  int col = blockIdx.x;
  __shared__ float sh[256];
  float mx = -3.402823e38f;
  for (int r = threadIdx.x; r < R; r += blockDim.x)
    mx = fmaxf(mx, LA[(size_t)r * C + col]);
  sh[threadIdx.x] = mx; __syncthreads();
  for (int w = 128; w > 0; w >>= 1) {
    if ((int)threadIdx.x < w) sh[threadIdx.x] = fmaxf(sh[threadIdx.x], sh[threadIdx.x + w]);
    __syncthreads();
  }
  mx = sh[0]; __syncthreads();
  float s = 0.f;
  for (int r = threadIdx.x; r < R; r += blockDim.x)
    s += expf(LA[(size_t)r * C + col] - mx);
  sh[threadIdx.x] = s; __syncthreads();
  for (int w = 128; w > 0; w >>= 1) {
    if ((int)threadIdx.x < w) sh[threadIdx.x] += sh[threadIdx.x + w];
    __syncthreads();
  }
  float lse = mx + logf(sh[0]);
  for (int r = threadIdx.x; r < R; r += blockDim.x)
    LA[(size_t)r * C + col] -= lse;
}

__global__ void k_exp_ip(float* p, long n) {
  long i = (long)blockIdx.x * blockDim.x + threadIdx.x;
  if (i < n) p[i] = expf(p[i]);
}

__global__ void k_colmean_tanh(const float* H, float* ctx, int N, int C) {
  int c = blockIdx.x;
  __shared__ float sh[256];
  float s = 0.f;
  for (int r = threadIdx.x; r < N; r += blockDim.x) s += H[(size_t)r * C + c];
  sh[threadIdx.x] = s; __syncthreads();
  for (int w = 128; w > 0; w >>= 1) {
    if ((int)threadIdx.x < w) sh[threadIdx.x] += sh[threadIdx.x + w];
    __syncthreads();
  }
  if (threadIdx.x == 0) ctx[c] = tanhf(sh[0] / (float)N);
}

__global__ void k_scores(const float* AF, const float* ctx, float* sc, int N, int C) {
  int r = blockIdx.x * blockDim.x + threadIdx.x;
  if (r >= N) return;
  const float* row = AF + (size_t)r * C;
  float s = 0.f;
  for (int c = 0; c < C; ++c) s += row[c] * ctx[c];
  sc[r] = 1.f / (1.f + expf(-s));
}

__global__ void k_pool(const float* AF, const float* sc, float* P, int N, int C) {
  int c = blockIdx.x;
  __shared__ float sh[256];
  float s = 0.f;
  for (int r = threadIdx.x; r < N; r += blockDim.x) s += AF[(size_t)r * C + c] * sc[r];
  sh[threadIdx.x] = s; __syncthreads();
  for (int w = 128; w > 0; w >>= 1) {
    if ((int)threadIdx.x < w) sh[threadIdx.x] += sh[threadIdx.x + w];
    __syncthreads();
  }
  if (threadIdx.x == 0) P[c] = sh[0];
}

// tensor-net + MLP head, single block of 64 threads
__global__ void k_head(const float* p1, const float* p2, const float* tnW,
                       const float* tnBlock, const float* tnBias,
                       const float* f1w, const float* f1b,
                       const float* f2w, const float* f2b,
                       const float* f3w, const float* f3b,
                       const float* sw, const float* sb, float* out) {
  __shared__ float q[256], s[32], a1[64], a2[32], a3[16];
  int t = threadIdx.x;
  for (int i = t; i < 256; i += blockDim.x) q[i] = (i < 128) ? p1[i] : p2[i - 128];
  __syncthreads();
  if (t < 32) {
    float acc = 0.f;
    for (int i = 0; i < 128; ++i) {
      const float* w = tnW + (size_t)i * 128 * 32 + t;
      float inner = 0.f;
      for (int j = 0; j < 128; ++j) inner += w[(size_t)j * 32] * q[128 + j];
      acc += q[i] * inner;
    }
    float blk = 0.f;
    for (int c = 0; c < 256; ++c) blk += tnBlock[t * 256 + c] * q[c];
    float val = acc + blk + tnBias[t];
    s[t] = val > 0.f ? val : 0.f;
  }
  __syncthreads();
  if (t < 64) { float a = f1b[t]; for (int i = 0; i < 32; ++i) a += s[i] * f1w[i * 64 + t];
                a1[t] = a > 0.f ? a : 0.f; }
  __syncthreads();
  if (t < 32) { float a = f2b[t]; for (int i = 0; i < 64; ++i) a += a1[i] * f2w[i * 32 + t];
                a2[t] = a > 0.f ? a : 0.f; }
  __syncthreads();
  if (t < 16) { float a = f3b[t]; for (int i = 0; i < 32; ++i) a += a2[i] * f3w[i * 16 + t];
                a3[t] = a > 0.f ? a : 0.f; }
  __syncthreads();
  if (t == 0) { float a = sb[0]; for (int i = 0; i < 16; ++i) a += a3[i] * sw[i];
                out[0] = a; }
}

// ---------------------------------------------------------------------------
// Host orchestration
// ---------------------------------------------------------------------------
extern "C" void kernel_launch(void* const* d_in, const int* in_sizes, int n_in,
                              void* d_out, int out_size, void* d_ws, size_t ws_size,
                              hipStream_t stream) {
  (void)in_sizes; (void)n_in; (void)out_size; (void)ws_size;
  auto F  = [&](int i) { return (const float*)d_in[i]; };
  auto I  = [&](int i) { return (const int*)d_in[i]; };

  // data inputs
  const float *feat1 = F(0), *feat2 = F(1), *lgf1 = F(2), *lgf2 = F(3);
  const int *ei1 = I(4), *ei2 = I(5), *lgei1 = I(6), *lgei2 = I(7);
  // params, JAX pytree (alphabetical dict) order
  const float* att_w = F(8);
  const float *costW = F(9),  *costV = F(10);
  const float *fc1b = F(11), *fc1w = F(12);
  const float *fc2b = F(13), *fc2w = F(14);
  const float *fc3b = F(15), *fc3w = F(16);
  const float *lgmW = F(59), *lgmV = F(60);
  const float *mapW = F(61), *mapV = F(62);
  const float *scoreB = F(63), *scoreW = F(64);
  const float *tnW = F(65), *tnBias = F(66), *tnBlock = F(67);

  struct Gin { const float *beta,*eps,*gamma,*l1b,*l1w,*l2b,*l2w; int din,dout; };
  int dims[3][2] = {{64,256},{256,128},{128,128}};
  Gin G[2][3];
  for (int s = 0; s < 2; ++s)
    for (int l = 0; l < 3; ++l) {
      int b = (s == 0 ? 17 : 38) + l * 7;
      G[s][l] = { F(b+0), F(b+1), F(b+2), F(b+3), F(b+4), F(b+5), F(b+6),
                  dims[l][0], dims[l][1] };
    }

  // ---- workspace layout ----
  char* ws = (char*)d_ws;
  size_t off = 0;
  auto alloc = [&](size_t bytes) -> char* {
    char* p = ws + off; off += (bytes + 255) & ~(size_t)255; return p;
  };
  float*  Xbuf  = (float*) alloc((size_t)3072*256*4);
  float*  AGG   = (float*) alloc((size_t)3072*256*4);
  float*  H1    = (float*) alloc((size_t)3072*256*4);
  float*  H2    = (float*) alloc((size_t)3072*256*4);
  __bf16* XBF   = (__bf16*)alloc((size_t)3072*256*2);
  __bf16* H1BF  = (__bf16*)alloc((size_t)3072*256*2);
  float*  STATS = (float*) alloc(512*4);
  float*  AF1   = (float*) alloc((size_t)2048*128*4);
  float*  AF2   = (float*) alloc((size_t)2048*128*4);
  float*  LAF1  = (float*) alloc((size_t)3072*128*4);
  float*  LAF2  = (float*) alloc((size_t)3072*128*4);
  __bf16* AF1B  = (__bf16*)alloc((size_t)2048*128*2);
  __bf16* AF2B  = (__bf16*)alloc((size_t)2048*128*2);
  __bf16* LAF1B = (__bf16*)alloc((size_t)3072*128*2);
  __bf16* LAF2B = (__bf16*)alloc((size_t)3072*128*2);
  __bf16* TKB   = (__bf16*)alloc((size_t)3072*2048*2);
  __bf16* WT1[2][3]; __bf16* WT2[2][3];
  for (int s = 0; s < 2; ++s)
    for (int l = 0; l < 3; ++l) {
      WT1[s][l] = (__bf16*)alloc((size_t)dims[l][0]*dims[l][1]*2);
      WT2[s][l] = (__bf16*)alloc((size_t)dims[l][1]*dims[l][1]*2);
    }
  __bf16* ATTWT  = (__bf16*)alloc((size_t)128*128*2);
  __bf16* MAPWT  = (__bf16*)alloc((size_t)16*128*128*2);
  __bf16* COSTWT = (__bf16*)alloc((size_t)16*128*128*2);
  __bf16* LGWT   = (__bf16*)alloc((size_t)16*128*128*2);
  float* CTX = (float*)alloc(128*4);
  float* SC  = (float*)alloc(3072*4);
  float* P1  = (float*)alloc(128*4);
  float* P2  = (float*)alloc(128*4);

  auto g1 = [](long n) { return dim3((unsigned)((n + 255) / 256)); };

  // ---- weight transposes (f32 -> bf16 NT operands) ----
  for (int s = 0; s < 2; ++s)
    for (int l = 0; l < 3; ++l) {
      long n1 = (long)dims[l][0]*dims[l][1];
      long n2 = (long)dims[l][1]*dims[l][1];
      k_transpose_bf16<<<g1(n1),256,0,stream>>>(G[s][l].l1w, WT1[s][l], dims[l][0], dims[l][1], 1);
      k_transpose_bf16<<<g1(n2),256,0,stream>>>(G[s][l].l2w, WT2[s][l], dims[l][1], dims[l][1], 1);
    }
  k_transpose_bf16<<<g1(128*128),256,0,stream>>>(att_w, ATTWT, 128, 128, 1);
  k_transpose_bf16<<<g1(16L*128*128),256,0,stream>>>(mapW,  MAPWT,  128, 128, 16);
  k_transpose_bf16<<<g1(16L*128*128),256,0,stream>>>(costW, COSTWT, 128, 128, 16);
  k_transpose_bf16<<<g1(16L*128*128),256,0,stream>>>(lgmW,  LGWT,   128, 128, 16);

  auto gemm = [&](const __bf16* A, const __bf16* B, const float* bias,
                  float* C, __bf16* Cbf, int M, int N, int K,
                  int lda, int ldb, int ldc, int act) {
    dim3 g((unsigned)((N + 127) / 128), (unsigned)((M + 63) / 64));
    k_gemm_nt<<<g, 256, 0, stream>>>(A, B, bias, C, Cbf, M, N, K, lda, ldb, ldc, act);
  };

  // ---- GIN conv pass (3 layers) ----
  auto run_conv = [&](const float* X0, const int* ei, int N, int E, int set,
                      float* AF, __bf16* AFB) {
    const float* xin = X0;
    for (int l = 0; l < 3; ++l) {
      int din = G[set][l].din, dout = G[set][l].dout;
      long nIn = (long)N * din, nOut = (long)N * dout;
      k_zero<<<g1(nIn),256,0,stream>>>(AGG, nIn);
      k_scatter<<<g1((long)E*din),256,0,stream>>>(xin, ei, AGG, E, din);
      k_combine<<<g1(nIn),256,0,stream>>>(xin, AGG, G[set][l].eps, XBF, nIn);
      gemm(XBF, WT1[set][l], G[set][l].l1b, H1, nullptr, N, dout, din, din, din, dout, 1);
      k_f32_to_bf16<<<g1(nOut),256,0,stream>>>(H1, H1BF, nOut);
      gemm(H1BF, WT2[set][l], G[set][l].l2b, H2, nullptr, N, dout, dout, dout, dout, dout, 0);
      k_bn_stats<<<dout,256,0,stream>>>(H2, STATS, N, dout);
      float* yout = (l < 2) ? Xbuf : AF;
      __bf16* ybf = (l < 2) ? nullptr : AFB;
      int act = (l < 2) ? 1 : 0;
      k_bn_apply<<<g1(nOut),256,0,stream>>>(H2, STATS, G[set][l].gamma, G[set][l].beta,
                                            yout, ybf, nOut, dout, 1.0f/(float)N, act);
      xin = yout;
    }
  };

  run_conv(feat1, ei1, 2048, 32768, 0, AF1, AF1B);
  run_conv(feat2, ei2, 2048, 32768, 0, AF2, AF2B);
  run_conv(lgf1, lgei1, 3072, 49152, 1, LAF1, LAF1B);
  run_conv(lgf2, lgei2, 3072, 49152, 1, LAF2, LAF2B);

  // ---- output regions ----
  float* OUT     = (float*)d_out;
  float* T_TRANS = OUT + 1;
  float* T_COST  = OUT + 1 + (size_t)2048*2048;
  float* T_LG    = OUT + 1 + (size_t)2*2048*2048;

  // ---- GedMatrix: cost, map (scaled for sinkhorn log-init), lg_map ----
  gemm(AF1B, COSTWT, nullptr, nullptr, TKB, 2048, 2048, 128, 128, 128, 2048, 0);
  k_ged<<<dim3(2048/128, 2048/64),256,0,stream>>>(TKB, AF2B, costV, T_COST, 2048, 2048, 1.0f);

  gemm(AF1B, MAPWT, nullptr, nullptr, TKB, 2048, 2048, 128, 128, 128, 2048, 0);
  k_ged<<<dim3(2048/128, 2048/64),256,0,stream>>>(TKB, AF2B, mapV, T_TRANS, 2048, 2048, 10.0f); // m / TAU

  gemm(LAF1B, LGWT, nullptr, nullptr, TKB, 3072, 2048, 128, 128, 128, 2048, 0);
  k_ged<<<dim3(3072/128, 3072/64),256,0,stream>>>(TKB, LAF2B, lgmV, T_LG, 3072, 3072, 1.0f);

  // ---- Sinkhorn (log-space, in place in the output transport region) ----
  for (int it = 0; it < 20; ++it) {
    k_lse_row<<<2048,256,0,stream>>>(T_TRANS, 2048, 2048);
    k_lse_col<<<2048,256,0,stream>>>(T_TRANS, 2048, 2048);
  }
  k_exp_ip<<<g1((long)2048*2048),256,0,stream>>>(T_TRANS, (long)2048*2048);

  // ---- attention pooling ----
  auto attpool = [&](const float* AF, const __bf16* AFB, int N, float* P) {
    gemm(AFB, ATTWT, nullptr, H1, nullptr, N, 128, 128, 128, 128, 128, 0);
    k_colmean_tanh<<<128,256,0,stream>>>(H1, CTX, N, 128);
    k_scores<<<g1(N),256,0,stream>>>(AF, CTX, SC, N, 128);
    k_pool<<<128,256,0,stream>>>(AF, SC, P, N, 128);
  };
  attpool(AF1, AF1B, 2048, P1);
  attpool(AF2, AF2B, 2048, P2);

  // ---- tensor network + MLP head -> score ----
  k_head<<<1,64,0,stream>>>(P1, P2, tnW, tnBlock, tnBias,
                            fc1w, fc1b, fc2w, fc2b, fc3w, fc3b,
                            scoreW, scoreB, OUT);
}
